// VisualizeInternalGates_60378650247501
// MI455X (gfx1250) — compile-verified, matrix operand order checked by reading, not run on hardware
//
#include <hip/hip_runtime.h>
#include <hip/hip_bf16.h>

#define T_STEPS 512
#define NBATCH  64
#define EMB     256
#define HID     512
#define KDIM    768   // HID + EMB
#define VOCAB   32000

typedef __attribute__((ext_vector_type(16))) _Float16 v16h;
typedef __attribute__((ext_vector_type(8)))  _Float16 v8h;
typedef __attribute__((ext_vector_type(4)))  _Float16 v4h;
typedef __attribute__((ext_vector_type(8)))  float    v8f;

// ---- WMMA helpers --------------------------------------------------------
// 16-bit A-matrix 16x32 striping (ISA 7.12.2): lane l holds row (l&15);
// elements 0..7 are K = (l>>4)*8 .. +7, elements 8..15 are K = 16+(l>>4)*8 .. +7.
// Same striping for B with W stored [N,K] row-major (W == B^T); any consistent
// K permutation across A and B leaves the dot product invariant.
__device__ __forceinline__ v16h load_frag_f16(const _Float16* row, int k0, int hi) {
  v8h lo = *(const v8h*)(row + k0 + hi * 8);
  v8h up = *(const v8h*)(row + k0 + 16 + hi * 8);
  v16h f;
#pragma unroll
  for (int i = 0; i < 8; ++i) { f[i] = lo[i]; f[i + 8] = up[i]; }
  return f;
}

__device__ __forceinline__ v16h load_frag_f32cvt(const float* row, int k0, int hi) {
  v16h f;
#pragma unroll
  for (int i = 0; i < 8; ++i) {
    f[i]     = (_Float16)row[k0 + hi * 8 + i];
    f[i + 8] = (_Float16)row[k0 + 16 + hi * 8 + i];
  }
  return f;
}

__device__ __forceinline__ v8f wmma_f16(v16h a, v16h b, v8f c) {
  // D = A*B + C, f32 accumulate: v_wmma_f32_16x16x32_f16
  return __builtin_amdgcn_wmma_f32_16x16x32_f16(false, a, false, b,
                                                (short)0, c, false, false);
}

// ---- Setup kernels -------------------------------------------------------

// X[t][b][e] = emb_table[tok[b][t]][e] -> f16.  One thread per 4 elements.
__global__ void __launch_bounds__(256)
embed_gather_f16(const int* __restrict__ toks, const float* __restrict__ emb,
                 _Float16* __restrict__ X) {
  int idx = blockIdx.x * blockDim.x + threadIdx.x;   // 2,097,152 total
  int e4 = (idx & 63) * 4;
  int b  = (idx >> 6) & 63;
  int t  = idx >> 12;
  if (t >= T_STEPS) return;
  int tok = toks[b * T_STEPS + t];
  float4 v = *(const float4*)(emb + (size_t)tok * EMB + e4);
  v4h o; o[0] = (_Float16)v.x; o[1] = (_Float16)v.y;
  o[2] = (_Float16)v.z; o[3] = (_Float16)v.w;
  *(v4h*)(X + ((size_t)t * NBATCH + b) * EMB + e4) = o;
}

// Convert GRU weights to f16; zero h (f32 master) and h16 (matmul copy).
__global__ void __launch_bounds__(256)
init_convert(const float* __restrict__ Wz, const float* __restrict__ Wr,
             const float* __restrict__ Wc, _Float16* __restrict__ Wz16,
             _Float16* __restrict__ Wr16, _Float16* __restrict__ Wc16,
             float* __restrict__ h, _Float16* __restrict__ h16) {
  int i = blockIdx.x * blockDim.x + threadIdx.x;
  const int NW = HID * KDIM;                         // 393,216
  const int NH = NBATCH * HID;                       // 32,768
  if (i < NW)                Wz16[i]          = (_Float16)Wz[i];
  else if (i < 2 * NW)       Wr16[i - NW]     = (_Float16)Wr[i - NW];
  else if (i < 3 * NW)       Wc16[i - 2 * NW] = (_Float16)Wc[i - 2 * NW];
  else if (i < 3 * NW + NH)  h[i - 3 * NW] = 0.0f;
  else if (i < 3 * NW + 2 * NH) h16[i - 3 * NW - NH] = (_Float16)0.0f;
}

// ---- Per-step GRU kernels (LDS-free: all operands f16 in cache) ----------

// Gates z,r: sigmoid([h16,x] @ {Wz,Wr}^T); write stacks + rh16 = r*h.
// 2 waves/block, 1 WMMA tile/wave; 256 tiles = 2 gates x 4 M x 32 N.
__global__ void __launch_bounds__(64)
gru_gates(int t, const _Float16* __restrict__ h16, const _Float16* __restrict__ X,
          const _Float16* __restrict__ Wz16, const _Float16* __restrict__ Wr16,
          float* __restrict__ zs, float* __restrict__ rs,
          _Float16* __restrict__ rh16) {
  const int tid  = threadIdx.x;
  const int lane = tid & 31;                        // wave32
  const int tile = blockIdx.x * 2 + (tid >> 5);     // 0..255
  const int g  = tile & 1;
  const int mt = (tile >> 1) & 3;                   // 4 M tiles (batch)
  const int nt = tile >> 3;                         // 32 N tiles (hidden)
  const int hi = lane >> 4, l16 = lane & 15;
  const int m  = mt * 16 + l16;
  const int n0 = nt * 16;

  const _Float16* arow_h = h16 + (size_t)m * HID;
  const _Float16* arow_x = X + ((size_t)t * NBATCH + m) * EMB;
  const _Float16* brow   = (g ? Wr16 : Wz16) + (size_t)(n0 + l16) * KDIM;

  v8f acc = {};
#pragma unroll
  for (int k0 = 0; k0 < KDIM; k0 += 32) {
    v16h a = (k0 < HID) ? load_frag_f16(arow_h, k0, hi)
                        : load_frag_f16(arow_x, k0 - HID, hi);
    v16h b = load_frag_f16(brow, k0, hi);
    acc = wmma_f16(a, b, acc);
  }

  float* ostack = g ? rs : zs;
#pragma unroll
  for (int j = 0; j < 8; ++j) {                     // D: m = j + 8*hi, n = l16
    int mr = mt * 16 + j + 8 * hi;
    int n  = n0 + l16;
    float s = 1.0f / (1.0f + __expf(-acc[j]));
    ostack[(size_t)t * (NBATCH * HID) + (size_t)mr * HID + n] = s;
    if (g) {
      float hv = (float)h16[(size_t)mr * HID + n];
      rh16[mr * HID + n] = (_Float16)(s * hv);
    }
  }
}

// Candidate + state: h~ = tanh([rh,x] @ W^T); h = (1-z)h + z*h~ (f32 + f16).
// 128 tiles = 4 M x 32 N; 2 waves/block.
__global__ void __launch_bounds__(64)
gru_cand(int t, const _Float16* __restrict__ rh16, const _Float16* __restrict__ X,
         const _Float16* __restrict__ Wc16, const float* __restrict__ zs,
         float* __restrict__ hts, float* __restrict__ h,
         _Float16* __restrict__ h16) {
  const int tid  = threadIdx.x;
  const int lane = tid & 31;
  const int tile = blockIdx.x * 2 + (tid >> 5);     // 0..127
  const int mt = tile & 3;
  const int nt = tile >> 2;                         // 0..31
  const int hi = lane >> 4, l16 = lane & 15;
  const int m  = mt * 16 + l16;
  const int n0 = nt * 16;

  const _Float16* arow_r = rh16 + (size_t)m * HID;
  const _Float16* arow_x = X + ((size_t)t * NBATCH + m) * EMB;
  const _Float16* brow   = Wc16 + (size_t)(n0 + l16) * KDIM;

  v8f acc = {};
#pragma unroll
  for (int k0 = 0; k0 < KDIM; k0 += 32) {
    v16h a = (k0 < HID) ? load_frag_f16(arow_r, k0, hi)
                        : load_frag_f16(arow_x, k0 - HID, hi);
    v16h b = load_frag_f16(brow, k0, hi);
    acc = wmma_f16(a, b, acc);
  }

#pragma unroll
  for (int j = 0; j < 8; ++j) {
    int mr = mt * 16 + j + 8 * hi;
    int n  = n0 + l16;
    float ht = tanhf(acc[j]);
    size_t gidx = (size_t)t * (NBATCH * HID) + (size_t)mr * HID + n;
    hts[gidx] = ht;
    float z = zs[gidx];
    int hidx = mr * HID + n;                        // each (m,n) owned by 1 lane
    float hnew = (1.0f - z) * h[hidx] + z * ht;
    h[hidx]   = hnew;
    h16[hidx] = (_Float16)hnew;
  }
}

// ---- Classifier: logits = h @ cls_w^T + b; stream f32, convert in regs ---
__global__ void __launch_bounds__(256)
classifier(const float* __restrict__ h, const float* __restrict__ cw,
           const float* __restrict__ cb, float* __restrict__ logits) {
  const int tid = threadIdx.x;
  const int w = tid >> 5, lane = tid & 31;
  const int mt  = w & 3;                            // 4 M tiles
  const int ntl = w >> 2;                           // 2 N tiles per block
  const int n0 = (blockIdx.x * 2 + ntl) * 16;
  const int hi = lane >> 4, l16 = lane & 15;
  const float* arow = h  + (size_t)(mt * 16 + l16) * HID;
  const float* brow = cw + (size_t)(n0 + l16) * HID;

  v8f acc = {};
  for (int k0 = 0; k0 < HID; k0 += 32) {
    v16h a = load_frag_f32cvt(arow, k0, hi);
    v16h b = load_frag_f32cvt(brow, k0, hi);
    acc = wmma_f16(a, b, acc);
  }
#pragma unroll
  for (int j = 0; j < 8; ++j) {
    int m = mt * 16 + j + 8 * hi;
    int n = n0 + l16;
    logits[(size_t)m * VOCAB + n] = acc[j] + cb[n];
  }
}

// ---- Host-side launch ----------------------------------------------------
extern "C" void kernel_launch(void* const* d_in, const int* in_sizes, int n_in,
                              void* d_out, int out_size, void* d_ws, size_t ws_size,
                              hipStream_t stream) {
  (void)in_sizes; (void)n_in; (void)out_size; (void)ws_size;
  const int*   toks = (const int*)  d_in[0];
  const float* emb  = (const float*)d_in[1];
  const float* Wz   = (const float*)d_in[2];
  const float* Wr   = (const float*)d_in[3];
  const float* Wc   = (const float*)d_in[4];
  const float* cw   = (const float*)d_in[5];
  const float* cb   = (const float*)d_in[6];

  char* ws = (char*)d_ws;
  _Float16* X    = (_Float16*)ws; ws += (size_t)T_STEPS * NBATCH * EMB * 2; // 16.78 MB
  _Float16* Wz16 = (_Float16*)ws; ws += (size_t)HID * KDIM * 2;
  _Float16* Wr16 = (_Float16*)ws; ws += (size_t)HID * KDIM * 2;
  _Float16* Wc16 = (_Float16*)ws; ws += (size_t)HID * KDIM * 2;
  float*    hbuf = (float*)ws;    ws += (size_t)NBATCH * HID * 4;
  _Float16* h16  = (_Float16*)ws; ws += (size_t)NBATCH * HID * 2;
  _Float16* rh16 = (_Float16*)ws; ws += (size_t)NBATCH * HID * 2;

  float* out    = (float*)d_out;               // (logits, zs, rs, hts) flat
  float* logits = out;
  float* zs  = out + (size_t)NBATCH * VOCAB;
  float* rs  = zs  + (size_t)T_STEPS * NBATCH * HID;
  float* hts = rs  + (size_t)T_STEPS * NBATCH * HID;

  embed_gather_f16<<<8192, 256, 0, stream>>>(toks, emb, X);
  init_convert<<<4864, 256, 0, stream>>>(Wz, Wr, Wc, Wz16, Wr16, Wc16, hbuf, h16);
  for (int t = 0; t < T_STEPS; ++t) {
    gru_gates<<<128, 64, 0, stream>>>(t, h16, X, Wz16, Wr16, zs, rs, rh16);
    gru_cand <<<64, 64, 0, stream>>>(t, rh16, X, Wc16, zs, hts, hbuf, h16);
  }
  classifier<<<1000, 256, 0, stream>>>(hbuf, cw, cb, logits);
}
